// FiberBundleAttention_10857677324484
// MI455X (gfx1250) — compile-verified, matrix operand order checked by reading, not run on hardware
//
#include <hip/hip_runtime.h>
#include <math.h>

typedef __attribute__((ext_vector_type(16))) _Float16 v16h;
typedef __attribute__((ext_vector_type(2)))  _Float16 v2h;
typedef __attribute__((ext_vector_type(8)))  float    v8f;
typedef __attribute__((ext_vector_type(4)))  float    v4f;
typedef __attribute__((ext_vector_type(2)))  float    v2f;

#define DL 128
#define DM 16
#define SO 120
#define SS 128
#define BB 4

// Branch-free tanh: 1 - 2/(exp(2x)+1).  exp overflow -> inf -> rcp -> 0 -> +1;
// exp underflow -> 0 -> 1-2 = -1.  No EXEC-mask divergence.
__device__ __forceinline__ float fast_tanh(float x) {
  float e = __expf(2.0f * x);
  return 1.0f - 2.0f * __builtin_amdgcn_rcpf(e + 1.0f);
}

// ---------------------------------------------------------------------------
// Kernel 0: prepack W1 (256x128) and W2 (120x256 padded to 128) into f16
// WMMA B-fragment order: packed[((kstep*NT + ntile)*32 + lane)*16 + t]
// ---------------------------------------------------------------------------
__global__ __launch_bounds__(256) void pack_weights_kernel(const float* __restrict__ W1,
                                                           const float* __restrict__ W2,
                                                           _Float16* __restrict__ w1p,
                                                           _Float16* __restrict__ w2p) {
  int idx = blockIdx.x * 256 + threadIdx.x;       // 0 .. 65535
  int t    = idx & 15;
  int lane = (idx >> 4) & 31;
  int rest = idx >> 9;                            // 0..63 -> W1, 64..127 -> W2
  int base = (lane >> 4) * 8;
  int v = t >> 1;
  int klocal = (v < 4) ? (base + 2 * v + (t & 1))
                       : (base + 16 + 2 * (v - 4) + (t & 1));
  if (rest < 64) {                                // W1: 4 ksteps x 16 ntiles
    int ks = rest >> 4, nt = rest & 15;
    int n = nt * 16 + (lane & 15);
    int k = ks * 32 + klocal;
    w1p[idx] = (_Float16)W1[n * DL + k];
  } else {                                        // W2: 8 ksteps x 8 ntiles
    int r2 = rest - 64;
    int ks = r2 >> 3, nt = r2 & 7;
    int n = nt * 16 + (lane & 15);
    int k = ks * 32 + klocal;
    float val = (n < SO) ? W2[n * 256 + k] : 0.0f;
    w2p[idx - 32768] = (_Float16)val;
  }
}

// ---------------------------------------------------------------------------
// Kernel 1: Q = x @ Wq^T + bq ; K = x @ Wk^T + bk.   One block per (b,s).
// ---------------------------------------------------------------------------
__global__ __launch_bounds__(128) void proj_qk_kernel(const float* __restrict__ xl,
                                                      const float* __restrict__ Wq,
                                                      const float* __restrict__ bq,
                                                      const float* __restrict__ Wk,
                                                      const float* __restrict__ bk,
                                                      float* __restrict__ Q,
                                                      float* __restrict__ K) {
  __shared__ float xr[DL];
  const int row = blockIdx.x;         // b*S + s
  const int d = threadIdx.x;
  xr[d] = xl[(size_t)row * DL + d];
  __syncthreads();
  float aq = bq[d], ak = bk[d];
  const float* wq = Wq + (size_t)d * DL;
  const float* wk = Wk + (size_t)d * DL;
  #pragma unroll 8
  for (int k = 0; k < DL; ++k) {
    float x = xr[k];
    aq = fmaf(x, wq[k], aq);
    ak = fmaf(x, wk[k], ak);
  }
  Q[(size_t)row * DL + d] = aq;
  K[(size_t)row * DL + d] = ak;
}

// ---------------------------------------------------------------------------
// Kernel 2: attn[b,j,i] = softmax_i( dot(Q[b,j], K[b,i]) / sqrt(128) ).
// ---------------------------------------------------------------------------
__global__ __launch_bounds__(128) void attn_kernel(const float* __restrict__ Q,
                                                   const float* __restrict__ K,
                                                   float* __restrict__ attn) {
  __shared__ float qr[DL];
  __shared__ float red[DL];
  const int bj = blockIdx.x;
  const int bt = bj >> 7;
  const int i = threadIdx.x;
  qr[i] = Q[(size_t)bj * DL + i];
  __syncthreads();
  float s = 0.f;
  const float* krow = K + ((size_t)bt * SS + i) * DL;
  #pragma unroll 8
  for (int k = 0; k < DL; ++k) s = fmaf(qr[k], krow[k], s);
  s *= 0.08838834764831845f;  // 1/sqrt(128)
  red[i] = s;
  __syncthreads();
  for (int off = 64; off > 0; off >>= 1) {
    if (i < off) red[i] = fmaxf(red[i], red[i + off]);
    __syncthreads();
  }
  float mx = red[0];
  __syncthreads();
  float e = __expf(s - mx);
  red[i] = e;
  __syncthreads();
  for (int off = 64; off > 0; off >>= 1) {
    if (i < off) red[i] += red[i + off];
    __syncthreads();
  }
  attn[(size_t)bj * SS + i] = e * (1.0f / red[0]);
}

// ---------------------------------------------------------------------------
// Kernel 3: fused fiber-bundle transport.
// One block per (b,j); 4 waves; each wave handles two 16-pair i-tiles.
// delta -> (WMMA f16) h -> (WMMA f16) lie -> skew -> expm with two
// interleaved V_WMMA_F32_16X16X4_F32 chains -> coalesced T_all, settle.
// ---------------------------------------------------------------------------
__global__ __launch_bounds__(128) void fiber_kernel(const float* __restrict__ xl,
                                                    const float* __restrict__ xmem,
                                                    const float* __restrict__ b1,
                                                    const float* __restrict__ b2,
                                                    const float* __restrict__ Wo,
                                                    const float* __restrict__ bo,
                                                    const _Float16* __restrict__ w1p,
                                                    const _Float16* __restrict__ w2p,
                                                    const float* __restrict__ attn,
                                                    float* __restrict__ out,
                                                    float* __restrict__ Tall) {
  __shared__ float xj[DL];                               // 512 B
  __shared__ float settled[4][16];                       // 256 B
  __shared__ float stot[16];                             // 64 B
  __shared__ __align__(16) _Float16 wbuf[4][4096];       // 32 KB: xi(f32) -> h(f16) -> lie(f16)
  __shared__ __align__(16) float Sb[4][256];             // chain-0 skew
  __shared__ __align__(16) float Pb[4][256];             // chain-0 power
  __shared__ __align__(16) float Sc[4][256];             // chain-1 skew
  __shared__ __align__(16) float Pc[4][256];             // chain-1 power
  __shared__ __align__(16) float xmt[4][256];            // x_memory tiles

  const int tid = threadIdx.x;
  const int w = tid >> 5;
  const int lane = tid & 31;
  const int bj = blockIdx.x;        // b*S + j
  const int bt = bj >> 7;
  const int m16 = lane & 15;
  const int hi = lane >> 4;
  const int base = hi * 8;

  xj[tid] = xl[(size_t)bj * DL + tid];
  if (lane < 16) settled[w][lane] = 0.f;
  __syncthreads();

  // 16x16 f32 matmul on wave-private LDS tiles via V_WMMA_F32_16X16X4_F32
  auto wmat = [&](const float* X, const float* Y) -> v8f {
    v8f c = {};
    #pragma unroll
    for (int kk = 0; kk < 4; ++kk) {
      int k0 = kk * 4 + 2 * hi;
      v2f a = *(const v2f*)(X + m16 * 16 + k0);
      v2f bfr;
      bfr[0] = Y[k0 * 16 + m16];
      bfr[1] = Y[(k0 + 1) * 16 + m16];
      c = __builtin_amdgcn_wmma_f32_16x16x4_f32(false, a, false, bfr, (short)0, c,
                                                false, false);
    }
    return c;
  };

  #pragma unroll 1
  for (int halfTile = 0; halfTile < 2; ++halfTile) {
    const int i0 = (w * 2 + halfTile) * 16;

    // --- stage xi tile (16 rows x 128 f32) into wave-private LDS, 128b ops ---
    float* xiw = (float*)wbuf[w];
    #pragma unroll
    for (int t = 0; t < 16; ++t) {
      int f4 = t * 32 + lane;                 // float4 index, 0..511
      int r = f4 >> 5, c4 = (f4 & 31) * 4;
      v4f d = *(const v4f*)(xl + ((size_t)bt * SS + i0 + r) * DL + c4);
      *((v4f*)xiw + f4) = d;
    }
    // --- stage x_memory tile (16 x 16 f32) ---
    #pragma unroll
    for (int t = 0; t < 2; ++t) {
      int f4 = t * 32 + lane;                 // float4 index, 0..63
      int r = f4 >> 2, c4 = (f4 & 3) * 4;
      v4f d = *(const v4f*)(xmem + ((size_t)bt * SS + i0 + r) * DM + c4);
      *((v4f*)xmt[w] + f4) = d;
    }

    // --- build A1 fragments (delta = xj - xi, f16, 16x32 per kstep) ---
    v16h a1[4];
    #pragma unroll
    for (int ks = 0; ks < 4; ++ks) {
      #pragma unroll
      for (int u = 0; u < 8; ++u) {
        int kl = (u < 4) ? (base + 2 * u) : (base + 16 + 2 * (u - 4));
        int k = ks * 32 + kl;
        v2f xi2 = *(const v2f*)(xiw + m16 * DL + k);
        v2f xj2 = *(const v2f*)(xj + k);
        a1[ks][2 * u]     = (_Float16)(xj2[0] - xi2[0]);
        a1[ks][2 * u + 1] = (_Float16)(xj2[1] - xi2[1]);
      }
    }

    // --- GEMM1: h = tanh(delta @ W1^T + b1) -> wave LDS (f16 16x256) ---
    _Float16* hb = wbuf[w];
    #pragma unroll 1
    for (int nt = 0; nt < 16; ++nt) {
      v8f c = {};
      #pragma unroll
      for (int ks = 0; ks < 4; ++ks) {
        v16h bfr = *(const v16h*)(w1p + (((size_t)(ks * 16 + nt)) * 32 + lane) * 16);
        c = __builtin_amdgcn_wmma_f32_16x16x32_f16(false, a1[ks], false, bfr,
                                                   (short)0, c, false, false);
      }
      int n = nt * 16 + m16;
      float bn = b1[n];
      #pragma unroll
      for (int v = 0; v < 8; ++v) {
        int mm = v + 8 * hi;
        hb[mm * 256 + n] = (_Float16)fast_tanh(c[v] + bn);
      }
    }

    // --- build A2 fragments from h (f16, 16x32 per kstep), paired loads ---
    v16h a2[8];
    #pragma unroll
    for (int ks = 0; ks < 8; ++ks) {
      #pragma unroll
      for (int u = 0; u < 8; ++u) {
        int kl = (u < 4) ? (base + 2 * u) : (base + 16 + 2 * (u - 4));
        v2h hp = *(const v2h*)(hb + m16 * 256 + ks * 32 + kl);
        a2[ks][2 * u]     = hp[0];
        a2[ks][2 * u + 1] = hp[1];
      }
    }

    // --- GEMM2: lie = h @ W2^T + b2 -> overlaid into wbuf (f16 16x120) ---
    _Float16* lb = wbuf[w];   // h fully consumed into a2 registers above
    #pragma unroll 1
    for (int nt = 0; nt < 8; ++nt) {
      v8f c = {};
      #pragma unroll
      for (int ks = 0; ks < 8; ++ks) {
        v16h bfr = *(const v16h*)(w2p + (((size_t)(ks * 8 + nt)) * 32 + lane) * 16);
        c = __builtin_amdgcn_wmma_f32_16x16x32_f16(false, a2[ks], false, bfr,
                                                   (short)0, c, false, false);
      }
      int n = nt * 16 + m16;
      if (n < SO) {
        float bn = b2[n];
        #pragma unroll
        for (int v = 0; v < 8; ++v) {
          int mm = v + 8 * hi;
          lb[mm * SO + n] = (_Float16)(c[v] + bn);
        }
      }
    }

    // --- expm: two interleaved chains (pairs p and p+1) ---
    const float invs = 1.0f / 64.0f;          // 2^-6 scaling
    #pragma unroll 1
    for (int p = 0; p < 16; p += 2) {
      // build scaled skew matrices
      #pragma unroll
      for (int t = 0; t < 8; ++t) {
        int fl = t * 32 + lane;
        int d = fl >> 4, e = fl & 15;
        float v0 = 0.f, v1 = 0.f;
        if (d < e) {
          int n = d * (31 - d) / 2 + (e - d - 1);
          v0 = invs * (float)lb[p * SO + n];
          v1 = invs * (float)lb[(p + 1) * SO + n];
        } else if (d > e) {
          int n = e * (31 - e) / 2 + (d - e - 1);
          v0 = -invs * (float)lb[p * SO + n];
          v1 = -invs * (float)lb[(p + 1) * SO + n];
        }
        Sb[w][fl] = v0;
        Sc[w][fl] = v1;
      }

      // T = I + A for both chains
      float tr0[8], tr1[8];
      #pragma unroll
      for (int v = 0; v < 8; ++v) {
        int mm = v + 8 * hi;
        float id = (mm == m16) ? 1.f : 0.f;
        tr0[v] = id + Sb[w][mm * 16 + m16];
        tr1[v] = id + Sc[w][mm * 16 + m16];
      }
      // P2 = A@A / 2
      {
        v8f c0 = wmat(Sb[w], Sb[w]);
        v8f c1 = wmat(Sc[w], Sc[w]);
        #pragma unroll
        for (int v = 0; v < 8; ++v) {
          int mm = (v + 8 * hi) * 16 + m16;
          float p0 = c0[v] * 0.5f, p1 = c1[v] * 0.5f;
          tr0[v] += p0; Pb[w][mm] = p0;
          tr1[v] += p1; Pc[w][mm] = p1;
        }
      }
      // Pk = P(k-1)@A / k,  k = 3..10
      #pragma unroll 1
      for (int k = 3; k <= 10; ++k) {
        v8f c0 = wmat(Pb[w], Sb[w]);
        v8f c1 = wmat(Pc[w], Sc[w]);
        float invk = 1.0f / (float)k;
        #pragma unroll
        for (int v = 0; v < 8; ++v) {
          int mm = (v + 8 * hi) * 16 + m16;
          float p0 = c0[v] * invk, p1 = c1[v] * invk;
          tr0[v] += p0; Pb[w][mm] = p0;
          tr1[v] += p1; Pc[w][mm] = p1;
        }
      }
      // 6 squarings: T <- T@T (both chains)
      #pragma unroll 1
      for (int q = 0; q < 6; ++q) {
        #pragma unroll
        for (int v = 0; v < 8; ++v) {
          int mm = (v + 8 * hi) * 16 + m16;
          Pb[w][mm] = tr0[v];
          Pc[w][mm] = tr1[v];
        }
        v8f c0 = wmat(Pb[w], Pb[w]);
        v8f c1 = wmat(Pc[w], Pc[w]);
        #pragma unroll
        for (int v = 0; v < 8; ++v) { tr0[v] = c0[v]; tr1[v] = c1[v]; }
      }

      // park final T in LDS (used for both the coalesced store and transport)
      #pragma unroll
      for (int v = 0; v < 8; ++v) {
        int mm = (v + 8 * hi) * 16 + m16;
        Pb[w][mm] = tr0[v];
        Pc[w][mm] = tr1[v];
      }

      // coalesced T_all emission: lane-contiguous 128b stores
      size_t tb0 = ((size_t)bj * SS + (i0 + p)) * 256;
      #pragma unroll
      for (int t = 0; t < 2; ++t) {
        int f4 = t * 32 + lane;               // float4 index, 0..63
        v4f d0 = *((const v4f*)Pb[w] + f4);
        v4f d1 = *((const v4f*)Pc[w] + f4);
        *(v4f*)(Tall + tb0 + (size_t)f4 * 4) = d0;
        *(v4f*)(Tall + tb0 + 256 + (size_t)f4 * 4) = d1;
      }

      // transported = T @ x_memory[b,i]; attention-weighted settle
      float av0 = attn[(size_t)bj * SS + i0 + p];
      float av1 = attn[(size_t)bj * SS + i0 + p + 1];
      if (lane < 16) {
        float t0 = 0.f, t1 = 0.f;
        #pragma unroll
        for (int e = 0; e < 16; ++e) {
          t0 = fmaf(Pb[w][lane * 16 + e], xmt[w][p * 16 + e], t0);
          t1 = fmaf(Pc[w][lane * 16 + e], xmt[w][(p + 1) * 16 + e], t1);
        }
        settled[w][lane] += av0 * t0 + av1 * t1;
      }
    }  // pairs
  }  // halfTile

  __syncthreads();
  if (tid < 16) {
    float s = 0.f;
    #pragma unroll
    for (int ww = 0; ww < 4; ++ww) s += settled[ww][tid];
    stot[tid] = s;
  }
  __syncthreads();
  if (tid < 16) {
    float o = bo[tid];
    #pragma unroll
    for (int e = 0; e < 16; ++e) o = fmaf(Wo[tid * 16 + e], stot[e], o);
    out[(size_t)bj * DM + tid] = o;
  }
}

// ---------------------------------------------------------------------------
extern "C" void kernel_launch(void* const* d_in, const int* in_sizes, int n_in,
                              void* d_out, int out_size, void* d_ws, size_t ws_size,
                              hipStream_t stream) {
  const float* xl = (const float*)d_in[0];
  const float* xm = (const float*)d_in[1];
  const float* Wq = (const float*)d_in[2];
  const float* bq = (const float*)d_in[3];
  const float* Wk = (const float*)d_in[4];
  const float* bk = (const float*)d_in[5];
  const float* W1 = (const float*)d_in[6];
  const float* b1 = (const float*)d_in[7];
  const float* W2 = (const float*)d_in[8];
  const float* b2 = (const float*)d_in[9];
  const float* Wo = (const float*)d_in[10];
  const float* bo = (const float*)d_in[11];

  char* ws = (char*)d_ws;
  float*    Q    = (float*)(ws + 0);          // 256 KB
  float*    K    = (float*)(ws + 262144);     // 256 KB
  float*    attn = (float*)(ws + 524288);     // 256 KB
  _Float16* w1p  = (_Float16*)(ws + 786432);  // 64 KB
  _Float16* w2p  = (_Float16*)(ws + 851968);  // 64 KB

  float* out  = (float*)d_out;                // [B,S,16] = 8192 floats
  float* Tall = out + (size_t)BB * SS * DM;   // [B,S,S,16,16]

  pack_weights_kernel<<<256, 256, 0, stream>>>(W1, W2, w1p, w2p);
  proj_qk_kernel<<<BB * SS, 128, 0, stream>>>(xl, Wq, bq, Wk, bk, Q, K);
  attn_kernel<<<BB * SS, 128, 0, stream>>>(Q, K, attn);
  fiber_kernel<<<BB * SS, 128, 0, stream>>>(xl, xm, b1, b2, Wo, bo, w1p, w2p,
                                            attn, out, Tall);
}
